// GAT_47064251630392
// MI455X (gfx1250) — matrix operand order repaired and checked by reading, not compiled, not run on hardware
//
#include <hip/hip_runtime.h>

typedef __attribute__((ext_vector_type(16))) _Float16 v16h;
typedef __attribute__((ext_vector_type(8)))  _Float16 v8h;
typedef __attribute__((ext_vector_type(4)))  _Float16 v4h;
typedef __attribute__((ext_vector_type(8)))  float    v8f;

#define N_NODES_C 100000
#define N_EDGES_C 1600000
#define ET_C      (N_EDGES_C + N_NODES_C)   // edges + self-loops
#define F_IN_C    128
#define HID_C     128                        // HEADS1*CH1
#define HEADS1_C  8
#define CH1_C     16
#define NCLS_C    64
#define NEG_SLOPE_C 0.2f

// ---------- helpers ----------
__device__ __forceinline__ unsigned f2ord(float x) {
  unsigned b = __float_as_uint(x);
  return (b & 0x80000000u) ? ~b : (b | 0x80000000u);
}
__device__ __forceinline__ float ord2f(unsigned u) {
  return __uint_as_float((u & 0x80000000u) ? (u & 0x7FFFFFFFu) : ~u);
}
__device__ __forceinline__ float lrelu(float x) { return x > 0.f ? x : NEG_SLOPE_C * x; }
__device__ __forceinline__ void edge_sd(const int* __restrict__ ei, int t, int& s, int& d) {
  if (t < N_EDGES_C) { s = ei[t]; d = ei[N_EDGES_C + t]; }
  else               { s = t - N_EDGES_C; d = s; }
}

// ---------- utility kernels ----------
__global__ void zero_u32(unsigned* __restrict__ p, int n) {
  int i = blockIdx.x * blockDim.x + threadIdx.x;
  if (i < n) p[i] = 0u;
}

// Pack W[K x NCOL] (f32, row-major) into WMMA B-operand lane layout (f16):
//   Bp[(((ct*(K/32) + kb)*32 + lane)*16 + e]  =  W[kb*32 + (e/8)*16 + (lane/16)*8 + e%8][ct*16 + lane%16]
// so in the GEMM each lane reads its 16 halves as one contiguous 32B chunk.
template <int K, int NCOL>
__global__ void pack_B(const float* __restrict__ W, _Float16* __restrict__ Bp) {
  int i = blockIdx.x * blockDim.x + threadIdx.x;
  if (i >= K * NCOL) return;
  int e    = i & 15;
  int lane = (i >> 4) & 31;
  int rest = i >> 9;
  int kb   = rest % (K / 32);
  int ct   = rest / (K / 32);
  int k    = kb * 32 + (e >> 3) * 16 + (lane >> 4) * 8 + (e & 7);
  int col  = ct * 16 + (lane & 15);
  Bp[i] = (_Float16)W[(size_t)k * NCOL + col];
}

// ---------- WMMA GEMM: C[M x NCOL] = A[M x K] (f32) @ Bp (pre-packed f16) ----------
// block = 128 threads = 4 waves; each wave owns one 16-col tile.
// grid.x = M/16 row tiles, grid.y = NCOL/64 col groups.
template <int K, int NCOL>
__global__ void gemm_wmma(const float* __restrict__ A, const _Float16* __restrict__ Bp,
                          float* __restrict__ C) {
  __shared__ _Float16 As[16 * K];
  const int tid = threadIdx.x;
  const int rowbase = blockIdx.x * 16;

  // stage A tile (16 x K) into LDS as f16: float4 loads -> v4h stores
  const float4* A4 = (const float4*)(A + (size_t)rowbase * K);
  v4h* As4 = (v4h*)As;
  for (int i = tid; i < 16 * K / 4; i += 128) {
    float4 v = A4[i];
    v4h hh = { (_Float16)v.x, (_Float16)v.y, (_Float16)v.z, (_Float16)v.w };
    As4[i] = hh;
  }
  __syncthreads();

  const int wave = tid >> 5;
  const int lane = tid & 31;
  const int hsel = lane >> 4;       // which 16-lane half
  const int l16  = lane & 15;
  const int ct   = blockIdx.y * 4 + wave;   // 16-col tile index
  const v16h* B16 = (const v16h*)Bp;

  v8f c = {};
#pragma unroll
  for (int kb = 0; kb < K / 32; ++kb) {
    // A operand: lane's 16 halves are two contiguous 16B chunks in LDS
    const v8h* ap = (const v8h*)&As[l16 * K + kb * 32 + hsel * 8];
    v8h alo = ap[0];      // K offsets hsel*8 + 0..7
    v8h ahi = ap[2];      // K offsets hsel*8 + 16..23
    v16h a;
#pragma unroll
    for (int e = 0; e < 8; ++e) { a[e] = alo[e]; a[8 + e] = ahi[e]; }
    // B operand: one contiguous 32B chunk per lane
    v16h b = B16[(ct * (K / 32) + kb) * 32 + lane];
    c = __builtin_amdgcn_wmma_f32_16x16x32_f16(false, a, false, b, (short)0, c, false, false);
  }
#pragma unroll
  for (int r = 0; r < 8; ++r) {
    int m = r + hsel * 8;           // D layout: VGPR r -> M=r (lanes 0-15), M=8+r (lanes 16-31)
    C[(size_t)(rowbase + m) * NCOL + ct * 16 + l16] = c[r];
  }
}

// ---------- attention score dot products ----------
__global__ void rowdot1(const float* __restrict__ h, const float* __restrict__ asrc,
                        const float* __restrict__ adst, float* __restrict__ asn,
                        float* __restrict__ adn) {
  int i = blockIdx.x * blockDim.x + threadIdx.x;   // n*8 + head
  if (i >= N_NODES_C * HEADS1_C) return;
  int n = i >> 3, hd = i & 7;
  const float4* hv = (const float4*)(h + (size_t)n * HID_C + hd * CH1_C);
  const float4* sv = (const float4*)(asrc + hd * CH1_C);
  const float4* dv = (const float4*)(adst + hd * CH1_C);
  float s = 0.f, d = 0.f;
#pragma unroll
  for (int q = 0; q < 4; ++q) {
    float4 hh = hv[q], ss = sv[q], dd = dv[q];
    s += hh.x * ss.x + hh.y * ss.y + hh.z * ss.z + hh.w * ss.w;
    d += hh.x * dd.x + hh.y * dd.y + hh.z * dd.z + hh.w * dd.w;
  }
  asn[i] = s; adn[i] = d;
}

__global__ void rowdot2(const float* __restrict__ h, const float* __restrict__ asrc,
                        const float* __restrict__ adst, float* __restrict__ asn,
                        float* __restrict__ adn) {
  int n = blockIdx.x * blockDim.x + threadIdx.x;
  if (n >= N_NODES_C) return;
  const float4* hv = (const float4*)(h + (size_t)n * NCLS_C);
  const float4* sv = (const float4*)asrc;
  const float4* dv = (const float4*)adst;
  float s = 0.f, d = 0.f;
#pragma unroll
  for (int q = 0; q < 16; ++q) {
    float4 hh = hv[q], ss = sv[q], dd = dv[q];
    s += hh.x * ss.x + hh.y * ss.y + hh.z * ss.z + hh.w * ss.w;
    d += hh.x * dd.x + hh.y * dd.y + hh.z * dd.z + hh.w * dd.w;
  }
  asn[n] = s; adn[n] = d;
}

// ---------- edge passes ----------
template <int H>
__global__ void edge_max(const int* __restrict__ ei, const float* __restrict__ asn,
                         const float* __restrict__ adn, unsigned* __restrict__ emax) {
  int i = blockIdx.x * blockDim.x + threadIdx.x;
  if (i >= ET_C * H) return;
  int t = i / H, hd = i - t * H;
  int s, d; edge_sd(ei, t, s, d);
  float e = lrelu(asn[s * H + hd] + adn[d * H + hd]);
  atomicMax(&emax[d * H + hd], f2ord(e));
}

template <int H>
__global__ void edge_denom(const int* __restrict__ ei, const float* __restrict__ asn,
                           const float* __restrict__ adn, const unsigned* __restrict__ emax,
                           float* __restrict__ denom) {
  int i = blockIdx.x * blockDim.x + threadIdx.x;
  if (i >= ET_C * H) return;
  int t = i / H, hd = i - t * H;
  int s, d; edge_sd(ei, t, s, d);
  int id = d * H + hd;
  float e = lrelu(asn[s * H + hd] + adn[id]);
  atomicAdd(&denom[id], __expf(e - ord2f(emax[id])));
}

__global__ void edge_agg1(const int* __restrict__ ei, const float* __restrict__ asn,
                          const float* __restrict__ adn, const unsigned* __restrict__ emax,
                          const float* __restrict__ denom, const float* __restrict__ h,
                          float* __restrict__ out) {
  int i = blockIdx.x * blockDim.x + threadIdx.x;   // (edge, head)
  if (i >= ET_C * HEADS1_C) return;
  int t = i >> 3, hd = i & 7;
  int s, d; edge_sd(ei, t, s, d);
  int id = d * HEADS1_C + hd;
  float e = lrelu(asn[s * HEADS1_C + hd] + adn[id]);
  float alpha = __expf(e - ord2f(emax[id])) / denom[id];
  const float4* hv = (const float4*)(h + (size_t)s * HID_C + hd * CH1_C);
  float* op = out + (size_t)d * HID_C + hd * CH1_C;
#pragma unroll
  for (int q = 0; q < 4; ++q) {
    float4 v = hv[q];
    atomicAdd(op + q * 4 + 0, v.x * alpha);
    atomicAdd(op + q * 4 + 1, v.y * alpha);
    atomicAdd(op + q * 4 + 2, v.z * alpha);
    atomicAdd(op + q * 4 + 3, v.w * alpha);
  }
}

__global__ void edge_agg2(const int* __restrict__ ei, const float* __restrict__ asn,
                          const float* __restrict__ adn, const unsigned* __restrict__ emax,
                          const float* __restrict__ denom, const float* __restrict__ h,
                          float* __restrict__ out) {
  int i = blockIdx.x * blockDim.x + threadIdx.x;   // (edge, 16-ch group)
  if (i >= ET_C * 4) return;
  int t = i >> 2, q = i & 3;
  int s, d; edge_sd(ei, t, s, d);
  float e = lrelu(asn[s] + adn[d]);
  float alpha = __expf(e - ord2f(emax[d])) / denom[d];
  const float4* hv = (const float4*)(h + (size_t)s * NCLS_C + q * 16);
  float* op = out + (size_t)d * NCLS_C + q * 16;
#pragma unroll
  for (int r = 0; r < 4; ++r) {
    float4 v = hv[r];
    atomicAdd(op + r * 4 + 0, v.x * alpha);
    atomicAdd(op + r * 4 + 1, v.y * alpha);
    atomicAdd(op + r * 4 + 2, v.z * alpha);
    atomicAdd(op + r * 4 + 3, v.w * alpha);
  }
}

// ---------- epilogues ----------
__global__ void bias_elu(float* __restrict__ out, const float* __restrict__ b) {
  int i = blockIdx.x * blockDim.x + threadIdx.x;
  if (i >= N_NODES_C * HID_C) return;
  float v = out[i] + b[i & (HID_C - 1)];
  out[i] = v > 0.f ? v : (__expf(v) - 1.f);
}

__global__ void bias_logsoftmax(float* __restrict__ out, const float* __restrict__ b) {
  int n = blockIdx.x * blockDim.x + threadIdx.x;
  if (n >= N_NODES_C) return;
  float v[NCLS_C];
  const float4* ov = (const float4*)(out + (size_t)n * NCLS_C);
  const float4* bv = (const float4*)b;
#pragma unroll
  for (int q = 0; q < 16; ++q) {
    float4 a = ov[q], bb = bv[q];
    v[4 * q + 0] = a.x + bb.x; v[4 * q + 1] = a.y + bb.y;
    v[4 * q + 2] = a.z + bb.z; v[4 * q + 3] = a.w + bb.w;
  }
  float m = -3.0e38f;
#pragma unroll
  for (int c = 0; c < NCLS_C; ++c) m = fmaxf(m, v[c]);
  float ssum = 0.f;
#pragma unroll
  for (int c = 0; c < NCLS_C; ++c) ssum += __expf(v[c] - m);
  float lse = m + __logf(ssum);
  float4* wv = (float4*)(out + (size_t)n * NCLS_C);
#pragma unroll
  for (int q = 0; q < 16; ++q) {
    float4 r;
    r.x = v[4 * q + 0] - lse; r.y = v[4 * q + 1] - lse;
    r.z = v[4 * q + 2] - lse; r.w = v[4 * q + 3] - lse;
    wv[q] = r;
  }
}

// ---------- host launcher ----------
extern "C" void kernel_launch(void* const* d_in, const int* in_sizes, int n_in,
                              void* d_out, int out_size, void* d_ws, size_t ws_size,
                              hipStream_t stream) {
  (void)in_sizes; (void)n_in; (void)out_size; (void)ws_size;
  const float* x   = (const float*)d_in[0];
  const int*   ei  = (const int*)  d_in[1];
  const float* W1  = (const float*)d_in[2];
  const float* as1 = (const float*)d_in[3];
  const float* ad1 = (const float*)d_in[4];
  const float* b1  = (const float*)d_in[5];
  const float* W2  = (const float*)d_in[6];
  const float* as2 = (const float*)d_in[7];
  const float* ad2 = (const float*)d_in[8];
  const float* b2  = (const float*)d_in[9];
  float* out = (float*)d_out;

  char* ws = (char*)d_ws;
  size_t off = 0;
  auto alloc = [&](size_t bytes) -> void* {
    void* p = ws + off;
    off = (off + bytes + 255) & ~(size_t)255;
    return p;
  };
  _Float16* B1p  = (_Float16*)alloc((size_t)F_IN_C * HID_C * 2);
  _Float16* B2p  = (_Float16*)alloc((size_t)HID_C * NCLS_C * 2);
  float*    h1   = (float*)   alloc((size_t)N_NODES_C * HID_C * 4);
  float*    out1 = (float*)   alloc((size_t)N_NODES_C * HID_C * 4);
  float*    asn  = (float*)   alloc((size_t)N_NODES_C * HEADS1_C * 4);
  float*    adn  = (float*)   alloc((size_t)N_NODES_C * HEADS1_C * 4);
  unsigned* emax = (unsigned*)alloc((size_t)N_NODES_C * HEADS1_C * 4);
  float*    den  = (float*)   alloc((size_t)N_NODES_C * HEADS1_C * 4);
  float*    h2   = h1;  // layer-2 reuses layer-1 feature buffer

  const int B = 256;
  // weight conversion + pack into WMMA lane layout
  pack_B<128, 128><<<(F_IN_C * HID_C + B - 1) / B, B, 0, stream>>>(W1, B1p);
  pack_B<128, 64><<<(HID_C * NCLS_C + B - 1) / B, B, 0, stream>>>(W2, B2p);

  // ---- layer 1 ----
  gemm_wmma<128, 128><<<dim3(N_NODES_C / 16, 2), 128, 0, stream>>>(x, B1p, h1);
  rowdot1<<<(N_NODES_C * HEADS1_C + B - 1) / B, B, 0, stream>>>(h1, as1, ad1, asn, adn);
  zero_u32<<<(N_NODES_C * HEADS1_C + B - 1) / B, B, 0, stream>>>(emax, N_NODES_C * HEADS1_C);
  zero_u32<<<(N_NODES_C * HEADS1_C + B - 1) / B, B, 0, stream>>>((unsigned*)den, N_NODES_C * HEADS1_C);
  zero_u32<<<(N_NODES_C * HID_C + B - 1) / B, B, 0, stream>>>((unsigned*)out1, N_NODES_C * HID_C);
  edge_max<HEADS1_C><<<(ET_C * HEADS1_C + B - 1) / B, B, 0, stream>>>(ei, asn, adn, emax);
  edge_denom<HEADS1_C><<<(ET_C * HEADS1_C + B - 1) / B, B, 0, stream>>>(ei, asn, adn, emax, den);
  edge_agg1<<<(ET_C * HEADS1_C + B - 1) / B, B, 0, stream>>>(ei, asn, adn, emax, den, h1, out1);
  bias_elu<<<(N_NODES_C * HID_C + B - 1) / B, B, 0, stream>>>(out1, b1);

  // ---- layer 2 ----
  gemm_wmma<128, 64><<<dim3(N_NODES_C / 16, 1), 128, 0, stream>>>(out1, B2p, h2);
  rowdot2<<<(N_NODES_C + B - 1) / B, B, 0, stream>>>(h2, as2, ad2, asn, adn);
  zero_u32<<<(N_NODES_C + B - 1) / B, B, 0, stream>>>(emax, N_NODES_C);
  zero_u32<<<(N_NODES_C + B - 1) / B, B, 0, stream>>>((unsigned*)den, N_NODES_C);
  zero_u32<<<(N_NODES_C * NCLS_C + B - 1) / B, B, 0, stream>>>((unsigned*)out, N_NODES_C * NCLS_C);
  edge_max<1><<<(ET_C + B - 1) / B, B, 0, stream>>>(ei, asn, adn, emax);
  edge_denom<1><<<(ET_C + B - 1) / B, B, 0, stream>>>(ei, asn, adn, emax, den);
  edge_agg2<<<(ET_C * 4 + B - 1) / B, B, 0, stream>>>(ei, asn, adn, emax, den, h2, out);
  bias_logsoftmax<<<(N_NODES_C + B - 1) / B, B, 0, stream>>>(out, b2);
}
